// RNNCRFTagger_31774168056148
// MI455X (gfx1250) — matrix operand order repaired and checked by reading, not compiled, not run on hardware
//
#include <hip/hip_runtime.h>
#include <cstdint>
#include <cstddef>

// ---------------------------------------------------------------------------
// BiLSTM-CRF tagger for MI455X (gfx1250, wave32, WMMA bf16 -> f32 accumulate)
//   T=256, B=64, EMB=300(pad 320), HID=256, 4H=1024, NLAB=20(pad 32)
// ---------------------------------------------------------------------------

typedef __bf16 bf16;
typedef __attribute__((ext_vector_type(16))) __bf16        v16bf;
typedef __attribute__((ext_vector_type(8)))  float         v8f;
typedef __attribute__((ext_vector_type(4)))  unsigned int  u32x4; // trivially-copyable 16B chunk

#define T_LEN  256
#define B_SZ   64
#define EMB_D  300
#define EMB_P  320
#define HID    256
#define G4     1024
#define NLAB   20
#define NLAB_P 32
#define RNN_D  512
#define ROWS   (T_LEN * B_SZ)   // 16384

// 32-byte fragment: 16 bf16 values per lane for the 16x16x32 bf16 WMMA
union AB16 { v16bf v; u32x4 q[2]; };

static __device__ __forceinline__ v8f wmma_bf16(v16bf a, v16bf b, v8f c) {
  // v_wmma_f32_16x16x32_bf16  D = A(16x32) * B(32x16) + C(16x16 f32)
  return __builtin_amdgcn_wmma_f32_16x16x32_bf16(false, a, false, b, (short)0, c,
                                                 false, false);
}
static __device__ __forceinline__ float sigf(float x) {
  return 1.0f / (1.0f + __expf(-x));
}

// ---------------------------------------------------------------------------
// Weight convert / pad f32 -> bf16
// ---------------------------------------------------------------------------
__global__ void cvt_pad_bf16(const float* __restrict__ src, bf16* __restrict__ dst,
                             int srows, int scols, int drows, int dcols) {
  int n = drows * dcols;
  for (int i = blockIdx.x * blockDim.x + threadIdx.x; i < n; i += gridDim.x * blockDim.x) {
    int r = i / dcols, c = i % dcols;
    float v = (r < srows && c < scols) ? src[r * scols + c] : 0.0f;
    dst[i] = (bf16)v;
  }
}

// ---------------------------------------------------------------------------
// Embedding gather: x_bf[t*B+b][0..319] = bf16(emb[sentences[t][b]][:]), pad K
// ---------------------------------------------------------------------------
__global__ void gather_embed(const float* __restrict__ emb, const int* __restrict__ sent,
                             bf16* __restrict__ xbf) {
  const int row  = blockIdx.x;            // t*B + b
  const int word = sent[row];
  const float* e = emb + (size_t)word * EMB_D;
  __builtin_prefetch(e, 0, 1);            // -> global_prefetch_b8
  bf16* dst = xbf + (size_t)row * EMB_P;
  for (int c = threadIdx.x; c < EMB_P; c += blockDim.x)
    dst[c] = (c < EMB_D) ? (bf16)e[c] : (bf16)0.0f;
}

// ---------------------------------------------------------------------------
// Input projection (bias fused): xproj[dir][t][g][b] = sum_e Wih[g][e]*x[t][b][e] + bias[g]
//   A = Wih (M=g, K=e), B = x_t^T (K=e, N=b).  Output [g][b] so the recurrence
//   reads its 8 C-lane addends as two contiguous b128 loads.
// grid (32, T, 2), block 256 (8 waves, 1 tile each: 64 g-tiles x 4 b-tiles)
// ---------------------------------------------------------------------------
__global__ void input_proj(const bf16* __restrict__ wih, const bf16* __restrict__ xbf,
                           const float* __restrict__ bias_f, const float* __restrict__ bias_b,
                           float* __restrict__ xproj) {
  const int lane = threadIdx.x & 31, wid = threadIdx.x >> 5;
  const int lm = lane & 15, hi = lane >> 4;
  const int tile = blockIdx.x * 8 + wid;          // 0..255
  const int gt = tile >> 2, bt = tile & 3;
  const int t = blockIdx.y, dir = blockIdx.z;

  const bf16*  W    = wih + (size_t)dir * (G4 * EMB_P);
  const float* bias = dir ? bias_b : bias_f;
  float*       out  = xproj + ((size_t)dir * T_LEN + t) * (G4 * B_SZ);

  v8f acc = {};
  const bf16* arow = W   + (size_t)(gt * 16 + lm) * EMB_P + hi * 16;           // A: M=lm
  const bf16* brow = xbf + (size_t)(t * B_SZ + bt * 16 + lm) * EMB_P + hi * 16; // B: N=lm

  #pragma unroll
  for (int kk = 0; kk < EMB_P; kk += 32) {
    AB16 A, Bm;
    const u32x4* pa = (const u32x4*)(arow + kk);
    A.q[0] = pa[0]; A.q[1] = pa[1];          // K 16 contiguous per lane-half
    const u32x4* pb = (const u32x4*)(brow + kk);
    Bm.q[0] = pb[0]; Bm.q[1] = pb[1];        // K 16 contiguous per lane-half
    acc = wmma_bf16(A.v, Bm.v, acc);
  }
  #pragma unroll
  for (int v = 0; v < 8; v++) {
    int g = gt * 16 + v + hi * 8;            // C: M = v + 8*(lane>=16)
    int b = bt * 16 + lm;                    //    N = lane&15
    out[(size_t)g * B_SZ + b] = acc[v] + bias[g];
  }
}

// ---------------------------------------------------------------------------
// LSTM recurrence: persistent workgroup per direction (grid=2, block=1024=32 waves).
// Per step: gates(64x1024) = h(64x256 LDS) @ Whh^T(256x1024 L2) + xproj (global).
// Wave w: hidden-column tile j=w>>1 (16 cols) and batch half mh=w&1 (2 m-tiles)
//   -> 8 WMMA tiles covering gate types i,f,g,o at the SAME (b, h) coordinates,
//   so the nonlinearity + cell update stay in accumulator registers.
// ---------------------------------------------------------------------------
__global__ void lstm_recur(const bf16* __restrict__ whh, const float* __restrict__ xproj,
                           bf16* __restrict__ rnn) {
  const int dir  = blockIdx.x;
  const int tid  = threadIdx.x;
  const int lane = tid & 31, wid = tid >> 5;
  const int lm = lane & 15, hi = lane >> 4;
  const int j  = wid >> 1;                 // hidden-column tile 0..15
  const int mh = wid & 1;                  // batch half 0..1
  const int hcol = j * 16 + lm;

  __shared__ bf16 h_sh[B_SZ * HID];        // 32 KB of the 320 KB WGP LDS

  const bf16*  W       = whh   + (size_t)dir * (G4 * HID);
  const float* xp_base = xproj + (size_t)dir * T_LEN * (G4 * B_SZ);

  for (int i = tid; i < B_SZ * HID; i += blockDim.x) h_sh[i] = (bf16)0.0f;
  __syncthreads();

  float cst[2][8];                         // cell state, pinned to C-fragment lanes
  #pragma unroll
  for (int mi = 0; mi < 2; mi++)
    #pragma unroll
    for (int v = 0; v < 8; v++) cst[mi][v] = 0.0f;

  const v8f vzero = {};

  for (int s = 0; s < T_LEN; s++) {
    const int t = dir ? (T_LEN - 1 - s) : s;

    v8f acc[2][4];
    #pragma unroll
    for (int mi = 0; mi < 2; mi++)
      #pragma unroll
      for (int q = 0; q < 4; q++) acc[mi][q] = vzero;

    // h(LDS) @ Whh^T(global) : K = 256 -> 8 k-steps x 8 WMMA = 64 WMMA / wave / step
    #pragma unroll
    for (int kk = 0; kk < HID; kk += 32) {
      AB16 A[2];
      #pragma unroll
      for (int mi = 0; mi < 2; mi++) {
        int b = (mh * 2 + mi) * 16 + lm;                       // A: M = batch row
        const u32x4* p = (const u32x4*)&h_sh[b * HID + kk + hi * 16];
        A[mi].q[0] = p[0]; A[mi].q[1] = p[1];                  // ds_load_b128 x2
      }
      AB16 Bq[4];
      #pragma unroll
      for (int q = 0; q < 4; q++) {
        int g = q * HID + hcol;                                 // B: N = gate col
        const u32x4* p = (const u32x4*)(W + (size_t)g * HID + kk + hi * 16);
        Bq[q].q[0] = p[0]; Bq[q].q[1] = p[1];
      }
      #pragma unroll
      for (int mi = 0; mi < 2; mi++)
        #pragma unroll
        for (int q = 0; q < 4; q++)
          acc[mi][q] = wmma_bf16(A[mi].v, Bq[q].v, acc[mi][q]);
    }

    // activations + cell update, fully in-register (gate order: i, f, g, o)
    const float* xp = xp_base + (size_t)t * (G4 * B_SZ);
    float hn[2][8];
    #pragma unroll
    for (int mi = 0; mi < 2; mi++) {
      #pragma unroll
      for (int v = 0; v < 8; v++) {
        int b = (mh * 2 + mi) * 16 + v + hi * 8;
        float gi = acc[mi][0][v] + xp[(size_t)(0 * HID + hcol) * B_SZ + b];
        float gf = acc[mi][1][v] + xp[(size_t)(1 * HID + hcol) * B_SZ + b];
        float gg = acc[mi][2][v] + xp[(size_t)(2 * HID + hcol) * B_SZ + b];
        float go = acc[mi][3][v] + xp[(size_t)(3 * HID + hcol) * B_SZ + b];
        float c = sigf(gf) * cst[mi][v] + sigf(gi) * tanhf(gg);
        cst[mi][v] = c;
        float h = sigf(go) * tanhf(c);
        hn[mi][v] = h;
        // rnn_out[t][b][dir*256 + hcol]  (bf16, A-operand layout for emissions)
        rnn[(size_t)(t * B_SZ + b) * RNN_D + dir * HID + hcol] = (bf16)h;
      }
    }

    __syncthreads();   // all waves done READING h_sh for this step
    #pragma unroll
    for (int mi = 0; mi < 2; mi++)
      #pragma unroll
      for (int v = 0; v < 8; v++) {
        int b = (mh * 2 + mi) * 16 + v + hi * 8;
        h_sh[b * HID + hcol] = (bf16)hn[mi][v];
      }
    __syncthreads();   // new h visible before next step's fragments
  }
}

// ---------------------------------------------------------------------------
// Emissions: (16384 x 512) @ W_top^T (512 x 20, padded to 32) + b_top -> f32
// grid 256, block 256 (8 waves, 1 tile each; 1024 m-tiles x 2 n-tiles)
// ---------------------------------------------------------------------------
__global__ void emissions_gemm(const bf16* __restrict__ rnn, const bf16* __restrict__ wtop,
                               const float* __restrict__ btop, float* __restrict__ emis) {
  const int lane = threadIdx.x & 31, wid = threadIdx.x >> 5;
  const int lm = lane & 15, hi = lane >> 4;
  const int tile = blockIdx.x * 8 + wid;   // 0..2047
  const int mt = tile >> 1, nt = tile & 1;

  v8f acc = {};
  const bf16* arow = rnn  + (size_t)(mt * 16 + lm) * RNN_D + hi * 16;
  const bf16* brow = wtop + (size_t)(nt * 16 + lm) * RNN_D + hi * 16;
  #pragma unroll
  for (int kk = 0; kk < RNN_D; kk += 32) {
    AB16 A, Bm;
    const u32x4* pa = (const u32x4*)(arow + kk);
    A.q[0] = pa[0]; A.q[1] = pa[1];
    const u32x4* pb = (const u32x4*)(brow + kk);
    Bm.q[0] = pb[0]; Bm.q[1] = pb[1];
    acc = wmma_bf16(A.v, Bm.v, acc);
  }
  #pragma unroll
  for (int v = 0; v < 8; v++) {
    int row = mt * 16 + v + hi * 8;
    int col = nt * 16 + lm;
    if (col < NLAB)
      emis[(size_t)row * NLAB + col] = acc[v] + btop[col];
  }
}

// ---------------------------------------------------------------------------
// CRF: numerator score + forward algorithm (logsumexp scan), one block.
// ---------------------------------------------------------------------------
__global__ void crf_forward(const float* __restrict__ emis, const int* __restrict__ sent,
                            const int* __restrict__ tags, const float* __restrict__ start_t,
                            const float* __restrict__ end_t, const float* __restrict__ trans,
                            float* __restrict__ out) {
  __shared__ float aA[B_SZ * NLAB], aB[B_SZ * NLAB], res[B_SZ];
  __shared__ float tr[NLAB * NLAB], st[NLAB], et[NLAB];
  const int tid = threadIdx.x;

  for (int i = tid; i < NLAB * NLAB; i += blockDim.x) tr[i] = trans[i];
  if (tid < NLAB) { st[tid] = start_t[tid]; et[tid] = end_t[tid]; }
  __syncthreads();

  // ---- numerator (gold path score), one thread per batch element ----
  if (tid < B_SZ) {
    const int b = tid;
    int y0 = tags[b];
    float sc = st[y0] + emis[(size_t)b * NLAB + y0];
    int cnt = 1;                                  // mask[0] guaranteed true
    for (int t = 1; t < T_LEN; t++) {
      int yp = tags[(t - 1) * B_SZ + b];
      int yt = tags[t * B_SZ + b];
      if (sent[t * B_SZ + b] != 0) {
        sc += tr[yp * NLAB + yt] + emis[(size_t)(t * B_SZ + b) * NLAB + yt];
        cnt++;
      }
    }
    int last = tags[(cnt - 1) * B_SZ + b];        // tags[sum(mask)-1]
    res[b] = sc + et[last];
  }

  // ---- forward algorithm: alpha ping-pong in LDS ----
  for (int idx = tid; idx < B_SZ * NLAB; idx += blockDim.x) {
    int b = idx / NLAB, j = idx % NLAB;
    aA[idx] = st[j] + emis[(size_t)b * NLAB + j];
  }
  __syncthreads();

  float* cur = aA; float* nxt = aB;
  for (int t = 1; t < T_LEN; t++) {
    for (int idx = tid; idx < B_SZ * NLAB; idx += blockDim.x) {
      int b = idx / NLAB, j = idx % NLAB;
      float m = -3.4e38f;
      #pragma unroll
      for (int i = 0; i < NLAB; i++)
        m = fmaxf(m, cur[b * NLAB + i] + tr[i * NLAB + j]);
      float s = 0.0f;
      #pragma unroll
      for (int i = 0; i < NLAB; i++)
        s += __expf(cur[b * NLAB + i] + tr[i * NLAB + j] - m);
      float nv = m + __logf(s) + emis[(size_t)(t * B_SZ + b) * NLAB + j];
      nxt[idx] = (sent[t * B_SZ + b] != 0) ? nv : cur[idx];
    }
    __syncthreads();
    float* tmp = cur; cur = nxt; nxt = tmp;
  }

  // ---- denominator + reduction ----
  if (tid < B_SZ) {
    const int b = tid;
    float m = -3.4e38f;
    for (int jj = 0; jj < NLAB; jj++) m = fmaxf(m, cur[b * NLAB + jj] + et[jj]);
    float s = 0.0f;
    for (int jj = 0; jj < NLAB; jj++) s += __expf(cur[b * NLAB + jj] + et[jj] - m);
    res[b] -= (m + __logf(s));
  }
  __syncthreads();
  if (tid == 0) {
    float s = 0.0f;
    for (int b = 0; b < B_SZ; b++) s += res[b];
    out[0] = -s;                                   // negative log-likelihood
  }
}

// ---------------------------------------------------------------------------
// Launch. Workspace layout (~158 MiB):
//   x_bf[16384*320] | Wih_bf[2*1024*320] | Whh_bf[2*1024*256] | Wtop_bf[32*512]
//   | xproj f32[2*256*1024*64] | rnn bf16[16384*512] | emis f32[16384*20]
// ---------------------------------------------------------------------------
extern "C" void kernel_launch(void* const* d_in, const int* in_sizes, int n_in,
                              void* d_out, int out_size, void* d_ws, size_t ws_size,
                              hipStream_t stream) {
  (void)in_sizes; (void)n_in; (void)out_size; (void)ws_size;

  const float* emb     = (const float*)d_in[0];
  const float* Wih_f   = (const float*)d_in[1];
  const float* Whh_f   = (const float*)d_in[2];
  const float* b_f     = (const float*)d_in[3];
  const float* Wih_b   = (const float*)d_in[4];
  const float* Whh_b   = (const float*)d_in[5];
  const float* b_b     = (const float*)d_in[6];
  const float* W_top   = (const float*)d_in[7];
  const float* b_top   = (const float*)d_in[8];
  const float* start_t = (const float*)d_in[9];
  const float* end_t   = (const float*)d_in[10];
  const float* trans   = (const float*)d_in[11];
  const int*   sent    = (const int*)d_in[12];
  const int*   labels  = (const int*)d_in[13];

  bf16*  x_bf    = (bf16*)d_ws;
  bf16*  wih_bf  = x_bf   + (size_t)ROWS * EMB_P;
  bf16*  whh_bf  = wih_bf + (size_t)2 * G4 * EMB_P;
  bf16*  wtop_bf = whh_bf + (size_t)2 * G4 * HID;
  float* xproj   = (float*)(wtop_bf + (size_t)NLAB_P * RNN_D);
  bf16*  rnn     = (bf16*)(xproj + (size_t)2 * T_LEN * G4 * B_SZ);
  float* emis    = (float*)(rnn + (size_t)ROWS * RNN_D);

  cvt_pad_bf16<<<512, 256, 0, stream>>>(Wih_f, wih_bf,                       G4, EMB_D, G4, EMB_P);
  cvt_pad_bf16<<<512, 256, 0, stream>>>(Wih_b, wih_bf + (size_t)G4 * EMB_P,  G4, EMB_D, G4, EMB_P);
  cvt_pad_bf16<<<512, 256, 0, stream>>>(Whh_f, whh_bf,                       G4, HID,   G4, HID);
  cvt_pad_bf16<<<512, 256, 0, stream>>>(Whh_b, whh_bf + (size_t)G4 * HID,    G4, HID,   G4, HID);
  cvt_pad_bf16<<<64,  256, 0, stream>>>(W_top, wtop_bf,                      NLAB, RNN_D, NLAB_P, RNN_D);

  gather_embed<<<ROWS, 128, 0, stream>>>(emb, sent, x_bf);

  input_proj<<<dim3(32, T_LEN, 2), 256, 0, stream>>>(wih_bf, x_bf, b_f, b_b, xproj);

  lstm_recur<<<2, 1024, 0, stream>>>(whh_bf, xproj, rnn);

  emissions_gemm<<<256, 256, 0, stream>>>(rnn, wtop_bf, b_top, emis);

  crf_forward<<<1, 1024, 0, stream>>>(emis, sent, labels, start_t, end_t, trans,
                                      (float*)d_out);
}
